// AttentionLayer_GRU_LSTM_37890201486074
// MI455X (gfx1250) — matrix operand order repaired and checked by reading, not compiled, not run on hardware
//
#include <hip/hip_runtime.h>
#include <hip/hip_bf16.h>

// Problem constants (from the reference): B=16, N=2048, D_IN=D_OUT=512
#define BATCH 16
#define SEQ   2048
#define DIM   512
#define M_TOTAL (BATCH * SEQ)   // 32768 token rows

typedef __attribute__((ext_vector_type(16))) _Float16 v16h;
typedef __attribute__((ext_vector_type(8)))  _Float16 v8h;
typedef __attribute__((ext_vector_type(8)))  float    v8f;

// Matches the async builtin's parameter pointee type exactly (GCC-style vector).
typedef int async_i32x4 __attribute__((vector_size(16)));

#define WMMA_F32_F16(a, b, c) \
  __builtin_amdgcn_wmma_f32_16x16x32_f16(false, (a), false, (b), (short)0, (c), false, false)

// ---- CDNA5 async global->LDS path (guarded; falls back to direct copies) ----
#if defined(__has_builtin)
#if __has_builtin(__builtin_amdgcn_global_load_async_to_lds_b128) && \
    __has_builtin(__builtin_amdgcn_s_wait_asynccnt)
#define USE_ASYNC_LDS 1
#endif
#endif
#ifndef USE_ASYNC_LDS
#define USE_ASYNC_LDS 0
#endif

#if USE_ASYNC_LDS
#define ASYNC_CPY_B128(gsrc, ldst)                                          \
  __builtin_amdgcn_global_load_async_to_lds_b128(                           \
      (__attribute__((address_space(1))) async_i32x4*)(gsrc),               \
      (__attribute__((address_space(3))) async_i32x4*)(ldst), 0, 0)
#define ASYNC_WAIT() __builtin_amdgcn_s_wait_asynccnt(0)
#else
#define ASYNC_CPY_B128(gsrc, ldst) (*(v8h*)(ldst) = *(const v8h*)(gsrc))
#define ASYNC_WAIT() ((void)0)
#endif

static __device__ __forceinline__ v16h make_v16(v8h lo, v8h hi) {
  return __builtin_shufflevector(lo, hi, 0,1,2,3,4,5,6,7,8,9,10,11,12,13,14,15);
}

// A fragment (16x32 f16): lane l holds row (l&15), h = l>>4.
//   a[j] = A[row][16*h + j], j = 0..15
static __device__ __forceinline__ v16h load_a_frag(const _Float16* p, int h) {
  v8h lo = *(const v8h*)(p + 16 * h);
  v8h hi = *(const v8h*)(p + 16 * h + 8);
  return make_v16(lo, hi);
}
// B fragment (32x16 f16): lane l holds column (l&15); b[j] = B[16*h + j][n].
// p points at 16 contiguous halfs B[16h..16h+15][n] (k-contiguous storage).
static __device__ __forceinline__ v16h load_b_frag(const _Float16* p) {
  v8h lo = *(const v8h*)(p);
  v8h hi = *(const v8h*)(p + 8);
  return make_v16(lo, hi);
}

// ---------------------------------------------------------------------------
// Kernel 0: one-time W -> W^T (f16) conversion.  Wt[mtx][n][k], n,k in [0,512).
// ---------------------------------------------------------------------------
__global__ __launch_bounds__(256) void wtrans_kernel(
    const float* __restrict__ W0, const float* __restrict__ W1,
    const float* __restrict__ W2, _Float16* __restrict__ Wt)
{
  __shared__ _Float16 Ts[64][72];
  const int mtx = blockIdx.z;
  const float* W = (mtx == 0) ? W0 : (mtx == 1) ? W1 : W2;
  const int k0 = blockIdx.x * 64, n0 = blockIdx.y * 64;
  const int tid = threadIdx.x;
  #pragma unroll
  for (int c = 0; c < 4; ++c) {
    const int idx = tid + c * 256;       // 0..1023 float4 chunks
    const int r   = idx >> 4;            // k row 0..63
    const int c4  = (idx & 15) << 2;     // n col
    const float4 f = *(const float4*)(W + (size_t)(k0 + r) * DIM + n0 + c4);
    Ts[c4 + 0][r] = (_Float16)f.x;
    Ts[c4 + 1][r] = (_Float16)f.y;
    Ts[c4 + 2][r] = (_Float16)f.z;
    Ts[c4 + 3][r] = (_Float16)f.w;
  }
  __syncthreads();
  #pragma unroll
  for (int c = 0; c < 2; ++c) {
    const int idx = tid + c * 256;       // 0..511 v8h chunks
    const int n   = idx >> 3;
    const int c8  = (idx & 7) << 3;
    *(v8h*)(Wt + (size_t)mtx * DIM * DIM + (size_t)(n0 + n) * DIM + k0 + c8) =
        *(const v8h*)&Ts[n][c8];
  }
}

// ---------------------------------------------------------------------------
// Kernel 1: fused Q/K/V projection.  64(M) x 64(N) tile per 8-wave WG.
// Double-buffered k-loop: async W^T tiles + register-staged x conversion.
// Q, K stored row-major f16 [M_TOTAL][DIM]; V stored transposed Vt[d][m].
// ---------------------------------------------------------------------------
__global__ __launch_bounds__(256) void qkv_proj_kernel(
    const float* __restrict__ x, const _Float16* __restrict__ Wt,
    const float* __restrict__ bq, const float* __restrict__ bk,
    const float* __restrict__ bv,
    _Float16* __restrict__ Qh, _Float16* __restrict__ Kh, _Float16* __restrict__ Vt)
{
  __shared__ _Float16 Xs[2][64][40];      // double-buffered 64 x 32k (+pad)
  __shared__ _Float16 Wls[2][3][64][40];  // double-buffered [mtx][n][k] W^T tiles
  __shared__ _Float16 TsV[64][72];        // V-tile transpose staging

  const int tid  = threadIdx.x;
  const int lane = tid & 31;
  const int wv   = tid >> 5;
  const int l16  = lane & 15;
  const int h    = lane >> 4;
  const int rg   = wv & 3;   // rows 16*rg
  const int cg   = wv >> 2;  // cols 32*cg (two n-tiles)

  const int m0 = blockIdx.x * 64;
  const int n0 = blockIdx.y * 64;

  auto issue_w = [&](int buf, int kb) {
    #pragma unroll
    for (int c = 0; c < 3; ++c) {
      const int idx = tid + c * 256;     // 0..767 v8h chunks
      const int mtx = idx >> 8;
      const int rem = idx & 255;
      const int r   = rem >> 2;          // n row 0..63
      const int c8  = (rem & 3) << 3;    // k offset
      ASYNC_CPY_B128(Wt + (size_t)mtx * DIM * DIM + (size_t)(n0 + r) * DIM + kb + c8,
                     &Wls[buf][mtx][r][c8]);
    }
  };
  auto load_x = [&](int kb, float4& f0, float4& f1) {
    const int i0 = tid, i1 = tid + 256;
    f0 = *(const float4*)(x + (size_t)(m0 + (i0 >> 3)) * DIM + kb + ((i0 & 7) << 2));
    f1 = *(const float4*)(x + (size_t)(m0 + (i1 >> 3)) * DIM + kb + ((i1 & 7) << 2));
  };
  auto store_x = [&](int buf, const float4& f0, const float4& f1) {
    {
      const int r = tid >> 3, c4 = (tid & 7) << 2;
      Xs[buf][r][c4 + 0] = (_Float16)f0.x;
      Xs[buf][r][c4 + 1] = (_Float16)f0.y;
      Xs[buf][r][c4 + 2] = (_Float16)f0.z;
      Xs[buf][r][c4 + 3] = (_Float16)f0.w;
    }
    {
      const int i1 = tid + 256;
      const int r = i1 >> 3, c4 = (i1 & 7) << 2;
      Xs[buf][r][c4 + 0] = (_Float16)f1.x;
      Xs[buf][r][c4 + 1] = (_Float16)f1.y;
      Xs[buf][r][c4 + 2] = (_Float16)f1.z;
      Xs[buf][r][c4 + 3] = (_Float16)f1.w;
    }
  };

  const v8f zero8 = {0.f,0.f,0.f,0.f,0.f,0.f,0.f,0.f};
  v8f acc[3][2];
  #pragma unroll
  for (int i = 0; i < 3; ++i)
    #pragma unroll
    for (int j = 0; j < 2; ++j) acc[i][j] = zero8;

  // Prologue: stage tile 0
  issue_w(0, 0);
  {
    float4 f0, f1;
    load_x(0, f0, f1);
    store_x(0, f0, f1);
  }
  ASYNC_WAIT();
  __syncthreads();

  for (int kb = 0; kb < DIM; kb += 32) {
    const int cur = (kb >> 5) & 1;
    const bool has_next = (kb + 32) < DIM;
    float4 g0{}, g1{};
    if (has_next) {                      // start next tile while computing
      issue_w(cur ^ 1, kb + 32);
      load_x(kb + 32, g0, g1);
    }

    // Compute on current buffers: distinct registers for all 6 B-fragments
    const v16h afrag = load_a_frag(&Xs[cur][16 * rg + l16][0], h);
    v16h bfr[6];
    #pragma unroll
    for (int nt = 0; nt < 2; ++nt)
      #pragma unroll
      for (int mtx = 0; mtx < 3; ++mtx)
        bfr[nt * 3 + mtx] = load_b_frag(&Wls[cur][mtx][32 * cg + 16 * nt + l16][16 * h]);
    #pragma unroll
    for (int nt = 0; nt < 2; ++nt)
      #pragma unroll
      for (int mtx = 0; mtx < 3; ++mtx)
        acc[mtx][nt] = WMMA_F32_F16(afrag, bfr[nt * 3 + mtx], acc[mtx][nt]);

    if (has_next) store_x(cur ^ 1, g0, g1);
    ASYNC_WAIT();
    __syncthreads();
  }

  // Q, K: bias + direct row-major store
  const float* const bias2[2] = { bq, bk };
  _Float16* const outp2[2]    = { Qh, Kh };
  #pragma unroll
  for (int mtx = 0; mtx < 2; ++mtx) {
    #pragma unroll
    for (int nt = 0; nt < 2; ++nt) {
      const int n = n0 + 32 * cg + 16 * nt + l16;
      const float bb = bias2[mtx][n];
      #pragma unroll
      for (int r = 0; r < 8; ++r) {
        const int m = m0 + 16 * rg + 8 * h + r;
        outp2[mtx][(size_t)m * DIM + n] = (_Float16)(acc[mtx][nt][r] + bb);
      }
    }
  }
  // V: bias, transpose through LDS, coalesced store to Vt[d][m]
  #pragma unroll
  for (int nt = 0; nt < 2; ++nt) {
    const int n = 32 * cg + 16 * nt + l16;
    const float bb = bv[n0 + n];
    #pragma unroll
    for (int r = 0; r < 8; ++r) {
      const int m = 16 * rg + 8 * h + r;
      TsV[n][m] = (_Float16)(acc[2][nt][r] + bb);
    }
  }
  __syncthreads();
  #pragma unroll
  for (int c = 0; c < 2; ++c) {
    const int idx = tid + c * 256;       // 0..511 v8h chunks
    const int n   = idx >> 3;
    const int c8  = (idx & 7) << 3;
    *(v8h*)(Vt + (size_t)(n0 + n) * M_TOTAL + m0 + c8) = *(const v8h*)&TsV[n][c8];
  }
}

// ---------------------------------------------------------------------------
// K / V^T tile loader: 32 keys x 512 dims, async into LDS when available.
// ---------------------------------------------------------------------------
static __device__ __forceinline__ void load_kv_tile(
    const _Float16* __restrict__ Kh, const _Float16* __restrict__ Vt,
    _Float16* Ksb, _Float16* Vtsb, int b, int k0, int tid)
{
  const size_t kbase = ((size_t)b * SEQ + (size_t)k0) * DIM;
  #pragma unroll
  for (int c = 0; c < 8; ++c) {
    const int idx = tid + c * 256;       // 2048 v8h chunks
    const int r   = idx >> 6;            // key row 0..31
    const int c8  = (idx & 63) << 3;
    ASYNC_CPY_B128(Kh + kbase + (size_t)r * DIM + c8, Ksb + r * 520 + c8);
  }
  const size_t vcol = (size_t)b * SEQ + (size_t)k0;
  #pragma unroll
  for (int c = 0; c < 8; ++c) {
    const int idx = tid + c * 256;       // 2048 v8h chunks
    const int d   = idx >> 2;            // dim row 0..511
    const int c8  = (idx & 3) << 3;      // key offset 0..24
    ASYNC_CPY_B128(Vt + (size_t)d * M_TOTAL + vcol + c8, Vtsb + d * 40 + c8);
  }
}

// ---------------------------------------------------------------------------
// Kernel 2: flash attention (online softmax, no scaling — matches reference).
// WG: 64 query rows x DIM, 32-key tiles, double-buffered K/V^T in LDS.
// ---------------------------------------------------------------------------
__global__ __launch_bounds__(256) void flash_attn_kernel(
    const _Float16* __restrict__ Qh, const _Float16* __restrict__ Kh,
    const _Float16* __restrict__ Vt, float* __restrict__ out)
{
  extern __shared__ char smem_raw[];
  _Float16* Qs  = (_Float16*)smem_raw;       // 64 x 520
  _Float16* Ks  = Qs + 64 * 520;             // 2 x (32 x 520)
  _Float16* Vts = Ks + 2 * 32 * 520;         // 2 x (512 x 40)
  _Float16* Ps  = Vts + 2 * 512 * 40;        // 64 x 40
  float* S32    = (float*)(Ps + 64 * 40);    // 64 x 34
  float* mstate = S32 + 64 * 34;
  float* lstate = mstate + 64;
  float* fac    = lstate + 64;

  const int tid  = threadIdx.x;
  const int lane = tid & 31;
  const int wv   = tid >> 5;
  const int l16  = lane & 15;
  const int h    = lane >> 4;
  const int rg   = wv & 3;     // q-row group
  const int g2   = wv >> 2;    // key half (S) / 256-col group (PV)

  const int qb = blockIdx.x;
  const int b  = blockIdx.y;
  const size_t qbase = ((size_t)b * SEQ + (size_t)qb * 64) * DIM;

  // Prime pipeline: issue tile-0 K/V^T loads, then load Q while they fly.
  load_kv_tile(Kh, Vt, Ks, Vts, b, 0, tid);
  #pragma unroll
  for (int c = 0; c < 16; ++c) {
    const int idx = tid + c * 256;       // 4096 v8h chunks
    const int r   = idx >> 6;
    const int c8  = (idx & 63) << 3;
    *(v8h*)(Qs + r * 520 + c8) = *(const v8h*)(Qh + qbase + (size_t)r * DIM + c8);
  }
  if (tid < 64) { mstate[tid] = -3.0e38f; lstate[tid] = 0.0f; }

  const v8f zero8 = {0.f,0.f,0.f,0.f,0.f,0.f,0.f,0.f};
  v8f acc[16];
  #pragma unroll
  for (int i = 0; i < 16; ++i) acc[i] = zero8;

  ASYNC_WAIT();
  __syncthreads();

  const int NT = SEQ / 32;
  for (int kt = 0; kt < NT; ++kt) {
    const int cur = kt & 1;
    _Float16* Ksc = Ks  + cur * (32 * 520);
    _Float16* Vtc = Vts + cur * (512 * 40);
    // Prefetch next tile into the other buffer (its last readers finished
    // before the barrier that ended the previous iteration).
    if (kt + 1 < NT)
      load_kv_tile(Kh, Vt, Ks + (cur ^ 1) * (32 * 520),
                   Vts + (cur ^ 1) * (512 * 40), b, (kt + 1) * 32, tid);

    // --- S = Q.K^T : one 16x16 tile per wave (rows 16*rg, keys 16*g2) ---
    v8f s = zero8;
    #pragma unroll 4
    for (int db = 0; db < DIM; db += 32) {
      const v16h a  = load_a_frag(&Qs[(16 * rg + l16) * 520 + db], h);
      const v16h bb = load_b_frag(&Ksc[(16 * g2 + l16) * 520 + db + 16 * h]);
      s = WMMA_F32_F16(a, bb, s);
    }
    #pragma unroll
    for (int r = 0; r < 8; ++r)
      S32[(16 * rg + 8 * h + r) * 34 + 16 * g2 + l16] = s[r];
    __syncthreads();

    // --- online softmax stats: one thread per query row ---
    if (tid < 64) {
      const float* srow = &S32[tid * 34];
      const float m_old = mstate[tid];
      float tmax = m_old;
      for (int j = 0; j < 32; ++j) tmax = fmaxf(tmax, srow[j]);
      const float f = __expf(m_old - tmax);
      float sum = 0.f;
      _Float16* prow = &Ps[tid * 40];
      for (int j = 0; j < 32; ++j) {
        const float e = __expf(srow[j] - tmax);
        sum += e;
        prow[j] = (_Float16)e;
      }
      lstate[tid] = lstate[tid] * f + sum;
      mstate[tid] = tmax;
      fac[tid] = f;
    }
    __syncthreads();

    // --- O = O*fac + P.V : rows 16*rg, cols 256*g2..+256 ---
    float myfac[8];
    #pragma unroll
    for (int r = 0; r < 8; ++r) myfac[r] = fac[16 * rg + 8 * h + r];
    const v16h pa = load_a_frag(&Ps[(16 * rg + l16) * 40], h);
    #pragma unroll
    for (int ntc = 0; ntc < 16; ntc += 4) {
      v16h vb[4];                         // group-of-4 B-fragments: overlapped loads
      #pragma unroll
      for (int j = 0; j < 4; ++j) {
        const int n = 256 * g2 + 16 * (ntc + j) + l16;
        vb[j] = load_b_frag(&Vtc[n * 40 + 16 * h]);
      }
      #pragma unroll
      for (int j = 0; j < 4; ++j) {
        #pragma unroll
        for (int r = 0; r < 8; ++r) acc[ntc + j][r] *= myfac[r];
        acc[ntc + j] = WMMA_F32_F16(pa, vb[j], acc[ntc + j]);
      }
    }

    ASYNC_WAIT();        // next tile fully in LDS
    __syncthreads();     // ... and visible to all waves
  }

  float invl[8];
  #pragma unroll
  for (int r = 0; r < 8; ++r) invl[r] = 1.0f / lstate[16 * rg + 8 * h + r];
  #pragma unroll
  for (int nt = 0; nt < 16; ++nt) {
    const int n = 256 * g2 + 16 * nt + l16;
    #pragma unroll
    for (int r = 0; r < 8; ++r) {
      const size_t row = (size_t)b * SEQ + (size_t)qb * 64 + 16 * rg + 8 * h + r;
      out[row * DIM + n] = acc[nt][r] * invl[r];
    }
  }
}

// ---------------------------------------------------------------------------
extern "C" void kernel_launch(void* const* d_in, const int* in_sizes, int n_in,
                              void* d_out, int out_size, void* d_ws, size_t ws_size,
                              hipStream_t stream) {
  (void)in_sizes; (void)n_in; (void)out_size; (void)ws_size;

  const float* x  = (const float*)d_in[0];
  const float* Wq = (const float*)d_in[1];
  const float* bq = (const float*)d_in[2];
  const float* Wk = (const float*)d_in[3];
  const float* bk = (const float*)d_in[4];
  const float* Wv = (const float*)d_in[5];
  const float* bv = (const float*)d_in[6];

  // Workspace: Q, K row-major; V transposed; W^T.  (3*32MB + 1.5MB)
  _Float16* Qh = (_Float16*)d_ws;
  _Float16* Kh = Qh + (size_t)M_TOTAL * DIM;
  _Float16* Vt = Kh + (size_t)M_TOTAL * DIM;
  _Float16* Wt = Vt + (size_t)M_TOTAL * DIM;

  dim3 gw(DIM / 64, DIM / 64, 3);
  wtrans_kernel<<<gw, 256, 0, stream>>>(Wq, Wk, Wv, Wt);

  dim3 gproj(M_TOTAL / 64, DIM / 64);
  qkv_proj_kernel<<<gproj, 256, 0, stream>>>(x, Wt, bq, bk, bv, Qh, Kh, Vt);

  const size_t shmem =
      (size_t)(64 * 520 + 2 * 32 * 520 + 2 * 512 * 40 + 64 * 40) * sizeof(_Float16) +
      (size_t)(64 * 34 + 3 * 64) * sizeof(float);   // = 229,632 B (< 320 KB)
  (void)hipFuncSetAttribute(reinterpret_cast<const void*>(flash_attn_kernel),
                            hipFuncAttributeMaxDynamicSharedMemorySize, (int)shmem);

  dim3 gattn(SEQ / 64, BATCH);
  flash_attn_kernel<<<gattn, 256, shmem, stream>>>(Qh, Kh, Vt, (float*)d_out);
}
